// PointCloudDeformNet_19799799234739
// MI455X (gfx1250) — matrix operand order repaired and checked by reading, not compiled, not run on hardware
//
#include <hip/hip_runtime.h>
#include <hip/hip_bf16.h>

// ---------------------------------------------------------------------------
// CDNA5 (gfx1250, wave32) implementation of PointCloudDeformNet.
// All linear layers run through v_wmma_f32_16x16x32_f16 (f16 in, f32 accum),
// with ping-pong LDS double buffering (one barrier per K-step, global loads
// for tile t+1 overlapped with WMMA issue for tile t).
// ---------------------------------------------------------------------------

typedef __attribute__((ext_vector_type(16))) _Float16 v16h;
typedef __attribute__((ext_vector_type(8)))  _Float16 v8h;
typedef __attribute__((ext_vector_type(8)))  float    v8f;

#define TILE_M 128
#define TILE_N 64
#define TILE_K 32
#define KPAD   8   // 40 halves/row: 80B row stride, 16B aligned

static __device__ __forceinline__ v16h frag16(const _Float16* p0, const _Float16* p1)
{
    v8h lo = *(const v8h*)p0;
    v8h hi = *(const v8h*)p1;
    return __builtin_shufflevector(lo, hi, 0, 1, 2, 3, 4, 5, 6, 7,
                                           8, 9, 10, 11, 12, 13, 14, 15);
}

// ---------------------------------------------------------------------------
// Generic GEMM:  Y[M,N] = act( X[M,K] @ W^T + bias ),  W is [N,K] row-major f16.
// Block = 256 threads (8 waves).  Block tile 128(M) x 64(N), K-step 32.
// Wave w (wm = w&3, wn = w>>2) owns a 32x32 sub-tile = 4 accumulators,
// issuing 4 v_wmma per K-step.  Double-buffered LDS, 1 barrier per K-step.
// ---------------------------------------------------------------------------
__global__ __launch_bounds__(256)
void gemm_bias_act(const float* __restrict__ X, int ldx,
                   const _Float16* __restrict__ W,
                   const float* __restrict__ bias,
                   float* __restrict__ Y, int ldy,
                   int M, int N, int K, int relu)
{
    __shared__ __align__(16) _Float16 sA[2][TILE_M][TILE_K + KPAD];
    __shared__ __align__(16) _Float16 sB[2][TILE_N][TILE_K + KPAD];

    const int tid  = threadIdx.x;
    const int lane = tid & 31;
    const int wave = tid >> 5;
    const int wm   = wave & 3;   // M sub-tile group (rows wm*16 and wm*16+64)
    const int wn   = wave >> 2;  // N sub-tile group (cols wn*32 .. +31)
    const int m0   = blockIdx.y * TILE_M;
    const int n0   = blockIdx.x * TILE_N;

    // staging coordinates: thread -> (row sr + 8*ii, col sc)
    const int sc = tid & 31;
    const int sr = tid >> 5;

    // fragment coordinates
    const int ml  = wm * 16 + (lane & 15);
    const int nl0 = wn * 32 + (lane & 15);
    const int hi  = lane >> 4;

    v8f acc00 = {}, acc01 = {}, acc10 = {}, acc11 = {};

    const int Kfull  = K & ~(TILE_K - 1);
    const int nsteps = Kfull / TILE_K;             // full tiles
    const int ntiles = nsteps + ((Kfull < K) ? 1 : 0);

    const float*    xbase = X + (size_t)(m0 + sr) * ldx + sc;
    const _Float16* wbase = W + (size_t)(n0 + sr) * K + sc;

    float    ar[16];
    _Float16 br[8];

    // Issue global loads for tile at k0 into registers (no LDS traffic).
    auto load_full = [&](int k0) {
#pragma unroll
        for (int ii = 0; ii < 16; ++ii)
            ar[ii] = xbase[(size_t)(8 * ii) * ldx + k0];
#pragma unroll
        for (int ii = 0; ii < 8; ++ii)
            br[ii] = wbase[(size_t)(8 * ii) * K + k0];
    };
    auto load_rem = [&](int k0) {
        const int gk = k0 + sc;
#pragma unroll
        for (int ii = 0; ii < 16; ++ii)
            ar[ii] = (gk < K) ? X[(size_t)(m0 + sr + 8 * ii) * ldx + gk] : 0.f;
#pragma unroll
        for (int ii = 0; ii < 8; ++ii)
            br[ii] = (gk < K) ? W[(size_t)(n0 + sr + 8 * ii) * K + gk]
                              : (_Float16)0.f;
    };
    auto store_tile = [&](int buf) {
#pragma unroll
        for (int ii = 0; ii < 16; ++ii)
            sA[buf][sr + 8 * ii][sc] = (_Float16)ar[ii];
#pragma unroll
        for (int ii = 0; ii < 8; ++ii)
            sB[buf][sr + 8 * ii][sc] = br[ii];
    };

    // Prologue: stage tile 0 into buffer 0.
    if (nsteps > 0) load_full(0); else load_rem(0);
    store_tile(0);

    for (int t = 0; t < ntiles; ++t) {
        __syncthreads();   // staged data for tile t visible; buffer (t+1)&1 free
        const int buf = t & 1;

        // Prefetch tile t+1 into registers (overlaps with WMMA below).
        if (t + 1 < ntiles) {
            const int k0 = (t + 1) * TILE_K;
            if (t + 1 < nsteps) load_full(k0); else load_rem(k0);
        }

        // CDNA5 16-bit WMMA fragments (ISA 7.12.2): contiguous 8-half runs.
        // A lane(l): m=l&15, halves [8*hi..+7] and [16+8*hi..+7]
        // B lane(l): n=l&15, halves [8*hi..+7] and [16+8*hi..+7]
        v16h a0 = frag16(&sA[buf][ml][8 * hi],        &sA[buf][ml][16 + 8 * hi]);
        v16h a1 = frag16(&sA[buf][ml + 64][8 * hi],   &sA[buf][ml + 64][16 + 8 * hi]);
        v16h b0 = frag16(&sB[buf][nl0][8 * hi],       &sB[buf][nl0][16 + 8 * hi]);
        v16h b1 = frag16(&sB[buf][nl0 + 16][8 * hi],  &sB[buf][nl0 + 16][16 + 8 * hi]);

        acc00 = __builtin_amdgcn_wmma_f32_16x16x32_f16(false, a0, false, b0,
                                                       (short)0, acc00, false, false);
        acc01 = __builtin_amdgcn_wmma_f32_16x16x32_f16(false, a0, false, b1,
                                                       (short)0, acc01, false, false);
        acc10 = __builtin_amdgcn_wmma_f32_16x16x32_f16(false, a1, false, b0,
                                                       (short)0, acc10, false, false);
        acc11 = __builtin_amdgcn_wmma_f32_16x16x32_f16(false, a1, false, b1,
                                                       (short)0, acc11, false, false);

        // Store prefetched tile into the other buffer (safe after the barrier
        // above: all waves finished reading it in iteration t-1).
        if (t + 1 < ntiles) store_tile((t + 1) & 1);
    }

    // Epilogue.  C/D layout: VGPR r -> M=r (lanes 0-15), M=r+8 (lanes 16-31).
    const int gn0 = n0 + nl0;
    const float bv0 = bias[gn0];
    const float bv1 = bias[gn0 + 16];
#pragma unroll
    for (int r = 0; r < 8; ++r) {
        int gm0 = m0 + wm * 16 + r + 8 * hi;
        int gm1 = gm0 + 64;
        float v;
        v = acc00[r] + bv0; if (relu) v = fmaxf(v, 0.f);
        Y[(size_t)gm0 * ldy + gn0] = v;
        v = acc01[r] + bv1; if (relu) v = fmaxf(v, 0.f);
        Y[(size_t)gm0 * ldy + gn0 + 16] = v;
        v = acc10[r] + bv0; if (relu) v = fmaxf(v, 0.f);
        Y[(size_t)gm1 * ldy + gn0] = v;
        v = acc11[r] + bv1; if (relu) v = fmaxf(v, 0.f);
        Y[(size_t)gm1 * ldy + gn0 + 16] = v;
    }
}

// ---------------------------------------------------------------------------
// fp32 -> f16 weight conversion
// ---------------------------------------------------------------------------
__global__ void f32_to_f16(const float* __restrict__ src,
                           _Float16* __restrict__ dst, int n)
{
    int i = blockIdx.x * blockDim.x + threadIdx.x;
    if (i < n) dst[i] = (_Float16)src[i];
}

// ---------------------------------------------------------------------------
// Per-(b,c) image statistics: mean and sqrt(unbiased var + eps).
// ---------------------------------------------------------------------------
__global__ __launch_bounds__(256)
void img_stats(const float* __restrict__ feat,
               float* __restrict__ mean_out,
               float* __restrict__ scale_out, int HW)
{
    const float* p = feat + (size_t)blockIdx.x * HW;
    float s = 0.f, sq = 0.f;
    for (int i = threadIdx.x; i < HW; i += 256) {
        float v = p[i];
        s += v; sq += v * v;
    }
    __shared__ float ss[256], sqq[256];
    ss[threadIdx.x] = s; sqq[threadIdx.x] = sq;
    __syncthreads();
    for (int o = 128; o > 0; o >>= 1) {
        if (threadIdx.x < o) {
            ss[threadIdx.x]  += ss[threadIdx.x + o];
            sqq[threadIdx.x] += sqq[threadIdx.x + o];
        }
        __syncthreads();
    }
    if (threadIdx.x == 0) {
        float m   = ss[0] / (float)HW;
        float var = (sqq[0] - ss[0] * m) / (float)(HW - 1);
        mean_out[blockIdx.x]  = m;
        scale_out[blockIdx.x] = sqrtf(var + 1e-8f);
    }
}

// ---------------------------------------------------------------------------
// AdaIN: per-row normalize over C (ddof=1), re-modulate with image stats.
// One wave per row, 8 rows per block.
// ---------------------------------------------------------------------------
__global__ __launch_bounds__(256)
void adain(const float* __restrict__ pc,
           const float* __restrict__ imean,
           const float* __restrict__ iscale,
           float* __restrict__ out, int Npts, int C)
{
    const int row  = blockIdx.x * 8 + (threadIdx.x >> 5);
    const int lane = threadIdx.x & 31;
    const int b    = row / Npts;
    const float* x = pc + (size_t)row * C;

    float s = 0.f, sq = 0.f;
    for (int c = lane; c < C; c += 32) {
        float v = x[c];
        s += v; sq += v * v;
    }
#pragma unroll
    for (int off = 16; off > 0; off >>= 1) {
        s  += __shfl_xor(s,  off, 32);
        sq += __shfl_xor(sq, off, 32);
    }
    const float mean = s / (float)C;
    const float var  = (sq - s * mean) / (float)(C - 1);
    const float inv  = 1.f / sqrtf(var + 1e-8f);

    float* o         = out + (size_t)row * C;
    const float* im  = imean  + (size_t)b * C;
    const float* isc = iscale + (size_t)b * C;
    for (int c = lane; c < C; c += 32)
        o[c] = ((x[c] - mean) * inv + im[c]) * isc[c];
}

// ---------------------------------------------------------------------------
// Projection: per-point perspective projection + bilinear weights, 4 levels.
// ---------------------------------------------------------------------------
struct ProjPrep { int xi1, xi2, yi1, yi2; float w11, w21, w12, w22; };

__global__ void proj_prep_k(const float* __restrict__ points,
                            const float* __restrict__ proj_mat,
                            ProjPrep* __restrict__ out,  // [4][M]
                            int Npts, int M)
{
    int m = blockIdx.x * blockDim.x + threadIdx.x;
    if (m >= M) return;
    int b = m / Npts;
    float px = points[m * 3 + 0], py = points[m * 3 + 1], pz = points[m * 3 + 2];
    const float* P = proj_mat + (size_t)b * 16;
    float c0 = P[0]  * px + P[1]  * py + P[2]  * pz + P[3];
    float c1 = P[4]  * px + P[5]  * py + P[6]  * pz + P[7];
    float c3 = P[12] * px + P[13] * py + P[14] * pz + P[15];
    float x = (c0 / c3 + 1.f) * 0.5f;
    float y = 1.f - (c1 / c3 + 1.f) * 0.5f;
    const int HS[4] = {112, 56, 28, 14};
#pragma unroll
    for (int l = 0; l < 4; ++l) {
        float Wf = (float)HS[l];
        float x0 = fminf(fmaxf(x * Wf, 0.f), Wf - 1.f);
        float y0 = fminf(fmaxf(y * Wf, 0.f), Wf - 1.f);
        float x1f = floorf(x0), x2f = ceilf(x0);
        float y1f = floorf(y0), y2f = ceilf(y0);
        ProjPrep pp;
        pp.xi1 = (int)x1f; pp.xi2 = (int)x2f;
        pp.yi1 = (int)y1f; pp.yi2 = (int)y2f;
        pp.w11 = (x2f - x0) * (y2f - y0);
        pp.w21 = (x0 - x1f) * (y2f - y0);
        pp.w12 = (x2f - x0) * (y0 - y1f);
        pp.w22 = (x0 - x1f) * (y0 - y1f);
        out[(size_t)l * M + m] = pp;
    }
}

__global__ void proj_gather(const float* __restrict__ feat,  // [B,C,H,W]
                            const ProjPrep* __restrict__ prep,
                            float* __restrict__ out,         // concat + slice
                            int Npts, int M, int C, int H, int W, int ldo)
{
    size_t id = (size_t)blockIdx.x * blockDim.x + threadIdx.x;
    if (id >= (size_t)M * C) return;
    int m = (int)(id / C), c = (int)(id % C);
    int b = m / Npts;
    ProjPrep pp = prep[m];
    const float* f = feat + (size_t)(b * C + c) * H * W;
    float q11 = f[pp.yi1 * W + pp.xi1];
    float q21 = f[pp.yi1 * W + pp.xi2];
    float q12 = f[pp.yi2 * W + pp.xi1];
    float q22 = f[pp.yi2 * W + pp.xi2];
    out[(size_t)m * ldo + c] =
        pp.w11 * q11 + pp.w21 * q21 + pp.w12 * q12 + pp.w22 * q22;
}

__global__ void copy_points(const float* __restrict__ pts,
                            float* __restrict__ catb, int M, int ldo, int off)
{
    int m = blockIdx.x * blockDim.x + threadIdx.x;
    if (m >= M) return;
    catb[(size_t)m * ldo + off + 0] = pts[m * 3 + 0];
    catb[(size_t)m * ldo + off + 1] = pts[m * 3 + 1];
    catb[(size_t)m * ldo + off + 2] = pts[m * 3 + 2];
}

// ---------------------------------------------------------------------------
// Final head: 128 -> 3 (fp32) + residual add of points.
// ---------------------------------------------------------------------------
__global__ void final_head(const float* __restrict__ x3,
                           const float* __restrict__ w,    // [3,128]
                           const float* __restrict__ bias, // [3]
                           const float* __restrict__ points,
                           float* __restrict__ out, int M)
{
    int m = blockIdx.x * blockDim.x + threadIdx.x;
    if (m >= M) return;
    const float* xr = x3 + (size_t)m * 128;
#pragma unroll
    for (int o = 0; o < 3; ++o) {
        float s = bias[o];
        const float* wr = w + o * 128;
        for (int k = 0; k < 128; ++k) s += xr[k] * wr[k];
        out[m * 3 + o] = s + points[m * 3 + o];
    }
}

// ---------------------------------------------------------------------------
// Host orchestration
// ---------------------------------------------------------------------------
extern "C" void kernel_launch(void* const* d_in, const int* in_sizes, int n_in,
                              void* d_out, int out_size, void* d_ws, size_t ws_size,
                              hipStream_t stream)
{
    (void)in_sizes; (void)n_in; (void)out_size; (void)ws_size;
    const int Bb = 8, Npts = 8192, M = Bb * Npts;
    const int LD = 1923;

    const float* points   = (const float*)d_in[0];
    const float* feat[4]  = {(const float*)d_in[1], (const float*)d_in[2],
                             (const float*)d_in[3], (const float*)d_in[4]};
    const float* proj_mat = (const float*)d_in[5];

    const float* bw1[4] = {(const float*)d_in[6],  (const float*)d_in[12],
                           (const float*)d_in[18], (const float*)d_in[24]};
    const float* bb1[4] = {(const float*)d_in[7],  (const float*)d_in[13],
                           (const float*)d_in[19], (const float*)d_in[25]};
    const float* bw2[4] = {(const float*)d_in[8],  (const float*)d_in[14],
                           (const float*)d_in[20], (const float*)d_in[26]};
    const float* bb2[4] = {(const float*)d_in[9],  (const float*)d_in[15],
                           (const float*)d_in[21], (const float*)d_in[27]};
    const float* tw[4]  = {(const float*)d_in[10], (const float*)d_in[16],
                           (const float*)d_in[22], (const float*)d_in[28]};
    const float* tb[4]  = {(const float*)d_in[11], (const float*)d_in[17],
                           (const float*)d_in[23], (const float*)d_in[29]};
    const float* d1_w = (const float*)d_in[30]; const float* d1_b = (const float*)d_in[31];
    const float* d2_w = (const float*)d_in[32]; const float* d2_b = (const float*)d_in[33];
    const float* d3_w = (const float*)d_in[34]; const float* d3_b = (const float*)d_in[35];
    const float* d4_w = (const float*)d_in[36]; const float* d4_b = (const float*)d_in[37];

    const int C[4]    = {64, 128, 256, 512};
    const int HS[4]   = {112, 56, 28, 14};
    const int Cin[4]  = {3, 64, 128, 256};
    const int poff[4] = {960, 1024, 1152, 1408};
    const int foff[4] = {0, 64, 192, 448};

    char* base = (char*)d_ws;
    size_t curo = 0;
    auto alloc = [&](size_t bytes) -> void* {
        curo = (curo + 255) & ~(size_t)255;
        void* p = base + curo;
        curo += bytes;
        return p;
    };

    _Float16 *hw1[4], *hw2[4], *htw[4];
    for (int l = 0; l < 4; ++l) {
        hw1[l] = (_Float16*)alloc((size_t)C[l] * Cin[l] * 2);
        hw2[l] = (_Float16*)alloc((size_t)C[l] * C[l] * 2);
        htw[l] = (_Float16*)alloc((size_t)C[l] * C[l] * 2);
    }
    _Float16* hd1 = (_Float16*)alloc((size_t)512 * LD * 2);
    _Float16* hd2 = (_Float16*)alloc((size_t)256 * 512 * 2);
    _Float16* hd3 = (_Float16*)alloc((size_t)128 * 256 * 2);

    float *imean[4], *iscale[4];
    for (int l = 0; l < 4; ++l) {
        imean[l]  = (float*)alloc((size_t)Bb * C[l] * 4);
        iscale[l] = (float*)alloc((size_t)Bb * C[l] * 4);
    }
    ProjPrep* prep = (ProjPrep*)alloc((size_t)4 * M * sizeof(ProjPrep));
    float* pc[4];
    for (int l = 0; l < 4; ++l) pc[l] = (float*)alloc((size_t)M * C[l] * 4);
    float* tmp  = (float*)alloc((size_t)M * 512 * 4);
    float* nbuf = (float*)alloc((size_t)M * 512 * 4);
    float* catb = (float*)alloc((size_t)M * LD * 4);
    float* x1   = (float*)alloc((size_t)M * 512 * 4);
    float* x2   = (float*)alloc((size_t)M * 256 * 4);
    float* x3   = (float*)alloc((size_t)M * 128 * 4);

    auto cvt = [&](const float* s, _Float16* dch, int n) {
        f32_to_f16<<<dim3((n + 255) / 256), dim3(256), 0, stream>>>(s, dch, n);
    };
    for (int l = 0; l < 4; ++l) {
        cvt(bw1[l], hw1[l], C[l] * Cin[l]);
        cvt(bw2[l], hw2[l], C[l] * C[l]);
        cvt(tw[l],  htw[l], C[l] * C[l]);
    }
    cvt(d1_w, hd1, 512 * LD);
    cvt(d2_w, hd2, 256 * 512);
    cvt(d3_w, hd3, 128 * 256);

    for (int l = 0; l < 4; ++l)
        img_stats<<<dim3(Bb * C[l]), dim3(256), 0, stream>>>(
            feat[l], imean[l], iscale[l], HS[l] * HS[l]);

    proj_prep_k<<<dim3((M + 255) / 256), dim3(256), 0, stream>>>(
        points, proj_mat, prep, Npts, M);
    for (int l = 0; l < 4; ++l) {
        size_t tot = (size_t)M * C[l];
        proj_gather<<<dim3((unsigned)((tot + 255) / 256)), dim3(256), 0, stream>>>(
            feat[l], prep + (size_t)l * M, catb + poff[l], Npts, M, C[l], HS[l], HS[l], LD);
    }
    copy_points<<<dim3((M + 255) / 256), dim3(256), 0, stream>>>(
        points, catb, M, LD, 1920);

    auto gemm = [&](const float* X, int ldx, const _Float16* W, const float* bias,
                    float* Y, int ldy, int N, int K, int relu) {
        dim3 grid((N + TILE_N - 1) / TILE_N, M / TILE_M);
        gemm_bias_act<<<grid, dim3(256), 0, stream>>>(
            X, ldx, W, bias, Y, ldy, M, N, K, relu);
    };

    const float* cur_pc = points;
    for (int l = 0; l < 4; ++l) {
        gemm(cur_pc, Cin[l], hw1[l], bb1[l], tmp, C[l], C[l], Cin[l], 1);
        gemm(tmp, C[l], hw2[l], bb2[l], pc[l], C[l], C[l], C[l], 1);
        adain<<<dim3(M / 8), dim3(256), 0, stream>>>(
            pc[l], imean[l], iscale[l], nbuf, Npts, C[l]);
        gemm(nbuf, C[l], htw[l], tb[l], catb + foff[l], LD, C[l], C[l], 0);
        cur_pc = pc[l];
    }

    gemm(catb, LD, hd1, d1_b, x1, 512, 512, LD, 1);
    gemm(x1, 512, hd2, d2_b, x2, 256, 256, 512, 1);
    gemm(x2, 256, hd3, d3_b, x3, 128, 128, 256, 1);
    final_head<<<dim3((M + 255) / 256), dim3(256), 0, stream>>>(
        x3, d4_w, d4_b, points, (float*)d_out, M);
}
